// BertLayer_48163763257382
// MI455X (gfx1250) — matrix operand order repaired and checked by reading, not compiled, not run on hardware
//
#include <hip/hip_runtime.h>

// ---------------- types & helpers ----------------
typedef __attribute__((ext_vector_type(16))) __bf16 v16bf;
typedef __attribute__((ext_vector_type(8)))  float  v8f;

__device__ __forceinline__ unsigned short f2bf(float f) {
  unsigned int u = __float_as_uint(f);
  u += 0x7FFFu + ((u >> 16) & 1u);          // round-to-nearest-even
  return (unsigned short)(u >> 16);
}
__device__ __forceinline__ float bf2f(unsigned short h) {
  return __uint_as_float(((unsigned int)h) << 16);
}
__device__ __forceinline__ v8f zero8() { v8f z; for (int i = 0; i < 8; ++i) z[i] = 0.f; return z; }

union FragBF { uint4 u[2]; v16bf v; };

__device__ __forceinline__ v8f wmma_bf16(v16bf a, v16bf b, v8f c) {
  return __builtin_amdgcn_wmma_f32_16x16x32_bf16(false, a, false, b, (short)0, c, false, false);
}

// ---- async global->LDS copy (CDNA5 GLOBAL_LOAD_ASYNC_TO_LDS_B128, ASYNCcnt) ----
// Builtin signature (from hipcc diagnostic): param0 = 'int __vector_size__(16) __device__ *'
// i.e. pointer-to-int4-vector in the global (AS1) address space; param1 = LDS side (AS3).
typedef int v4i_gcc __attribute__((vector_size(16)));
typedef __attribute__((address_space(1))) v4i_gcc GV4;
typedef __attribute__((address_space(3))) v4i_gcc LV4;

#if __has_builtin(__builtin_amdgcn_global_load_async_to_lds_b128)
#define HAVE_ASYNC_LDS 1
#else
#define HAVE_ASYNC_LDS 0
#endif

__device__ __forceinline__ void async_copy_b128(void* lds, const void* g) {
#if HAVE_ASYNC_LDS
  __builtin_amdgcn_global_load_async_to_lds_b128((GV4*)g, (LV4*)lds, 0, 0);
#else
  *(uint4*)lds = *(const uint4*)g;
#endif
}
__device__ __forceinline__ void async_wait0() {
#if HAVE_ASYNC_LDS
#if __has_builtin(__builtin_amdgcn_s_wait_asynccnt)
  __builtin_amdgcn_s_wait_asynccnt(0);
#else
  asm volatile("s_wait_asynccnt 0x0" ::: "memory");
#endif
#endif
}

// ---------------- problem constants ----------------
#define BB 16
#define SS 512
#define HH 1024
#define II 4096
#define NHH 16
#define DHH 64
#define EE 8

// ---------------- tiled bf16-WMMA GEMM (BM=BN=128, BK=64) ----------------
// C[M,N] = A[M,K] * B[K,N] (+bias, +residual / GELU), fp32 accumulate.
// A fp32 (converted in LDS fill) or bf16 (async-copied raw). B fp32 global,
// converted to bf16 in-register before the ds_store (transposed tile Bt[n][k]).
enum { EPI_BF16_BIAS = 0, EPI_F32_BIAS_RES = 1, EPI_GELU_BF16 = 2 };

template <bool ABF16, int EPI>
__global__ __launch_bounds__(256) void gemm_wmma(
    const void* __restrict__ Av, const float* __restrict__ Bw,
    const float* __restrict__ bias, const float* __restrict__ resid,
    void* __restrict__ Cv, int M, int N, int K,
    const int* __restrict__ choice,
    long long aBatch, long long wExp, long long biasExp,
    long long rBatch, long long cBatch) {
  __shared__ __align__(16) unsigned short As[128 * 64];  // [m][k] bf16, 16KB
  __shared__ __align__(16) unsigned short Bs[128 * 64];  // [n][k] bf16, 16KB

  const int tid = threadIdx.x;
  const int z = blockIdx.z;
  const int e = choice ? choice[z] : 0;
  const float* Af = ABF16 ? (const float*)nullptr
                          : ((const float*)Av + (size_t)z * aBatch);
  const unsigned short* Ah = ABF16 ? ((const unsigned short*)Av + (size_t)z * aBatch)
                                   : (const unsigned short*)nullptr;
  const float* Bp = Bw + (size_t)e * wExp;
  const float* bp = bias + (size_t)e * biasExp;
  const float* rp = resid ? resid + (size_t)z * rBatch : (const float*)nullptr;
  float* Cf = (float*)Cv + (size_t)z * cBatch;
  unsigned short* Ch = (unsigned short*)Cv + (size_t)z * cBatch;

  const int n0 = blockIdx.x * 128;
  const int m0 = blockIdx.y * 128;
  const int wid = tid >> 5, lane = tid & 31;
  const int wm = wid & 1, wn = wid >> 1;  // 2 x 4 wave grid -> 64x32 per wave
  const int l15 = lane & 15;
  const bool lo = lane < 16;

  v8f acc[4][2];
  for (int i = 0; i < 4; ++i)
    for (int j = 0; j < 2; ++j) acc[i][j] = zero8();

  for (int kb = 0; kb < K; kb += 64) {
    if (kb + 64 < K) {  // hint next K-slab into cache (global_prefetch_b8)
      __builtin_prefetch(Bp + (size_t)(kb + 64 + (tid >> 4)) * N + n0, 0, 1);
    }
    __syncthreads();
    // ---- fill A tile (128x64)
    if (ABF16) {
      for (int it = 0; it < 4; ++it) {       // 1024 b128 slots
        int s = it * 256 + tid;
        int m = s >> 3, kk = (s & 7) << 3;
        async_copy_b128(&As[m * 64 + kk], Ah + (size_t)(m0 + m) * K + kb + kk);
      }
    } else {
      for (int it = 0; it < 8; ++it) {       // 2048 float4 slots
        int s = it * 256 + tid;
        int m = s >> 4, kk = (s & 15) << 2;
        float4 a4 = *(const float4*)(Af + (size_t)(m0 + m) * K + kb + kk);
        ushort4 hh;
        hh.x = f2bf(a4.x); hh.y = f2bf(a4.y); hh.z = f2bf(a4.z); hh.w = f2bf(a4.w);
        *(ushort4*)&As[m * 64 + kk] = hh;
      }
    }
    // ---- fill B tile transposed (Bt[n][k]), coalesced along n, convert to bf16
    for (int it = 0; it < 8; ++it) {         // 2048 float4 slots
      int s = it * 256 + tid;
      int k2 = s >> 5, nn = (s & 31) << 2;
      float4 b4 = *(const float4*)(Bp + (size_t)(kb + k2) * N + n0 + nn);
      Bs[(nn + 0) * 64 + k2] = f2bf(b4.x);
      Bs[(nn + 1) * 64 + k2] = f2bf(b4.y);
      Bs[(nn + 2) * 64 + k2] = f2bf(b4.z);
      Bs[(nn + 3) * 64 + k2] = f2bf(b4.w);
    }
    if (ABF16) async_wait0();
    __syncthreads();

    // ---- two K=32 steps per stage, fragments per documented wave32 layouts
    for (int ks2 = 0; ks2 < 2; ++ks2) {
      FragBF af[4], bfr[2];
      const int ac0 = ks2 * 32 + (lo ? 0 : 8);  // A: lanes0-15 K{0..7,16..23}; 16-31 K{8..15,24..31}
      for (int mi = 0; mi < 4; ++mi) {
        int row = wm * 64 + mi * 16 + l15;
        af[mi].u[0] = *(const uint4*)&As[row * 64 + ac0];
        af[mi].u[1] = *(const uint4*)&As[row * 64 + ac0 + 16];
      }
      const int bk0 = ks2 * 32 + (lo ? 0 : 8);  // B: lanes0-15 K{0..7,16..23}; 16-31 K{8..15,24..31}
      for (int ni = 0; ni < 2; ++ni) {
        int col = wn * 32 + ni * 16 + l15;
        bfr[ni].u[0] = *(const uint4*)&Bs[col * 64 + bk0];
        bfr[ni].u[1] = *(const uint4*)&Bs[col * 64 + bk0 + 16];
      }
      for (int mi = 0; mi < 4; ++mi)
        for (int ni = 0; ni < 2; ++ni)
          acc[mi][ni] = wmma_bf16(af[mi].v, bfr[ni].v, acc[mi][ni]);
    }
  }

  // ---- epilogue (C layout: vgpr i -> row i or i+8; N across 16-lane halves)
  for (int mi = 0; mi < 4; ++mi)
    for (int ni = 0; ni < 2; ++ni) {
      int gn = n0 + wn * 32 + ni * 16 + l15;
      float bv = bp[gn];
      for (int i = 0; i < 8; ++i) {
        int gm = m0 + wm * 64 + mi * 16 + i + (lo ? 0 : 8);
        size_t off = (size_t)gm * N + gn;
        float vvv = acc[mi][ni][i] + bv;
        if (EPI == EPI_F32_BIAS_RES) {
          Cf[off] = vvv + rp[off];
        } else if (EPI == EPI_GELU_BF16) {
          float g = 0.5f * vvv * (1.0f + erff(vvv * 0.70710678118f));
          Ch[off] = f2bf(g);
        } else {
          Ch[off] = f2bf(vvv);
        }
      }
    }
}

// ---------------- flash attention (bf16 WMMA, online softmax) ----------------
__global__ __launch_bounds__(256) void attn_kernel(
    const unsigned short* __restrict__ q, const unsigned short* __restrict__ k,
    const unsigned short* __restrict__ v, unsigned short* __restrict__ ctx) {
  __shared__ __align__(16) unsigned short Kt[32 * 64];      // [key][dh]
  __shared__ __align__(16) unsigned short Vt[64 * 32];      // [dh][key] transposed
  __shared__ __align__(16) unsigned short Pt[8 * 16 * 32];  // per-wave P tile

  const int tid = threadIdx.x, wid = tid >> 5, lane = tid & 31;
  const int l15 = lane & 15;
  const bool lo = lane < 16;
  const int bh = blockIdx.x;
  const int b = bh / NHH, h = bh % NHH;
  const int s0 = blockIdx.y * 128 + wid * 16;  // 16 q-rows per wave
  const size_t base = (size_t)b * SS * HH + (size_t)h * DHH;

  // Q fragments (K=64 -> 2 fragments of K=32), loaded once
  FragBF qa[2];
  {
    const unsigned short* qp = q + base + (size_t)(s0 + l15) * HH;
    for (int f = 0; f < 2; ++f) {
      int c0 = f * 32 + (lo ? 0 : 8);
      qa[f].u[0] = *(const uint4*)(qp + c0);
      qa[f].u[1] = *(const uint4*)(qp + c0 + 16);
    }
  }

  v8f acc[4];
  for (int i = 0; i < 4; ++i) acc[i] = zero8();
  float mrun[8], lrun[8];
  for (int i = 0; i < 8; ++i) { mrun[i] = -1e30f; lrun[i] = 0.f; }

  for (int j0 = 0; j0 < SS; j0 += 32) {
    __syncthreads();
    // stage K tile via async global->LDS (raw bf16 copy, ASYNCcnt path)
    {
      int key = tid >> 3, d8 = (tid & 7) << 3;  // 256 b128 slots
      async_copy_b128(&Kt[key * 64 + d8], k + base + (size_t)(j0 + key) * HH + d8);
    }
    // stage V transposed (element shuffle -> manual ds stores)
    for (int it = 0; it < 2; ++it) {
      int s = it * 256 + tid;  // 512 ushort4 slots
      int key = s >> 4, d4 = (s & 15) << 2;
      ushort4 v4 = *(const ushort4*)(v + base + (size_t)(j0 + key) * HH + d4);
      Vt[(d4 + 0) * 32 + key] = v4.x;
      Vt[(d4 + 1) * 32 + key] = v4.y;
      Vt[(d4 + 2) * 32 + key] = v4.z;
      Vt[(d4 + 3) * 32 + key] = v4.w;
    }
    async_wait0();
    __syncthreads();

    // two 16-key score tiles (S = Q * K^T / 8)
    v8f st[2];
    for (int t = 0; t < 2; ++t) {
      FragBF kb0, kb1;
      int key = t * 16 + l15;
      int d0 = lo ? 0 : 8;
      kb0.u[0] = *(const uint4*)&Kt[key * 64 + d0];
      kb0.u[1] = *(const uint4*)&Kt[key * 64 + d0 + 16];
      kb1.u[0] = *(const uint4*)&Kt[key * 64 + 32 + d0];
      kb1.u[1] = *(const uint4*)&Kt[key * 64 + 32 + d0 + 16];
      v8f s8 = zero8();
      s8 = wmma_bf16(qa[0].v, kb0.v, s8);
      s8 = wmma_bf16(qa[1].v, kb1.v, s8);
      for (int i = 0; i < 8; ++i) s8[i] *= 0.125f;  // 1/sqrt(64)
      st[t] = s8;
    }
    // online softmax (per-row reductions across 16-lane halves)
    float alpha[8];
    for (int i = 0; i < 8; ++i) {
      float mx = fmaxf(st[0][i], st[1][i]);
      for (int d = 1; d < 16; d <<= 1) mx = fmaxf(mx, __shfl_xor(mx, d, 32));
      float mnew = fmaxf(mrun[i], mx);
      alpha[i] = __expf(mrun[i] - mnew);
      float p0 = __expf(st[0][i] - mnew);
      float p1 = __expf(st[1][i] - mnew);
      st[0][i] = p0; st[1][i] = p1;
      float r = p0 + p1;
      for (int d = 1; d < 16; d <<= 1) r += __shfl_xor(r, d, 32);
      lrun[i] = lrun[i] * alpha[i] + r;
      mrun[i] = mnew;
    }
    for (int nt = 0; nt < 4; ++nt)
      for (int i = 0; i < 8; ++i) acc[nt][i] *= alpha[i];

    // P: C-layout -> per-wave LDS -> A-fragment layout
    unsigned short* pw = &Pt[wid * 512];
    for (int t = 0; t < 2; ++t)
      for (int i = 0; i < 8; ++i) {
        int row = i + (lo ? 0 : 8);
        pw[row * 32 + t * 16 + l15] = f2bf(st[t][i]);
      }
    FragBF pa;
    {
      int c0 = lo ? 0 : 8;
      pa.u[0] = *(const uint4*)&pw[l15 * 32 + c0];
      pa.u[1] = *(const uint4*)&pw[l15 * 32 + c0 + 16];
    }
    for (int nt = 0; nt < 4; ++nt) {
      FragBF vb;
      int dh = nt * 16 + l15;
      int ks = lo ? 0 : 8;
      vb.u[0] = *(const uint4*)&Vt[dh * 32 + ks];
      vb.u[1] = *(const uint4*)&Vt[dh * 32 + ks + 16];
      acc[nt] = wmma_bf16(pa.v, vb.v, acc[nt]);
    }
  }
  // write ctx (bf16) normalized by row sum
  for (int nt = 0; nt < 4; ++nt)
    for (int i = 0; i < 8; ++i) {
      int s = s0 + i + (lo ? 0 : 8);
      int col = nt * 16 + l15;
      ctx[base + (size_t)s * HH + col] = f2bf(acc[nt][i] / lrun[i]);
    }
}

// ---------------- fused LN1 -> attn_out, LN2 -> x_ln (bf16) ----------------
__global__ __launch_bounds__(256) void ln_kernel(
    const float* __restrict__ y, const float* __restrict__ g1,
    const float* __restrict__ b1, const float* __restrict__ g2,
    const float* __restrict__ b2, float* __restrict__ ao,
    unsigned short* __restrict__ xln) {
  __shared__ float rs_[256], rq[256];
  const int row = blockIdx.x, tid = threadIdx.x;
  const int c = tid * 4;
  float4 y4 = *(const float4*)(y + (size_t)row * HH + c);
  float s = y4.x + y4.y + y4.z + y4.w;
  float q2 = y4.x * y4.x + y4.y * y4.y + y4.z * y4.z + y4.w * y4.w;
  rs_[tid] = s; rq[tid] = q2;
  __syncthreads();
  for (int off = 128; off > 0; off >>= 1) {
    if (tid < off) { rs_[tid] += rs_[tid + off]; rq[tid] += rq[tid + off]; }
    __syncthreads();
  }
  float mu = rs_[0] * (1.f / HH);
  float var = rq[0] * (1.f / HH) - mu * mu;
  float rstd = rsqrtf(var + 1e-12f);
  __syncthreads();
  float4 g4 = *(const float4*)(g1 + c), bb4 = *(const float4*)(b1 + c);
  float a0 = (y4.x - mu) * rstd * g4.x + bb4.x;
  float a1 = (y4.y - mu) * rstd * g4.y + bb4.y;
  float a2 = (y4.z - mu) * rstd * g4.z + bb4.z;
  float a3 = (y4.w - mu) * rstd * g4.w + bb4.w;
  *(float4*)(ao + (size_t)row * HH + c) = make_float4(a0, a1, a2, a3);
  s = a0 + a1 + a2 + a3;
  q2 = a0 * a0 + a1 * a1 + a2 * a2 + a3 * a3;
  rs_[tid] = s; rq[tid] = q2;
  __syncthreads();
  for (int off = 128; off > 0; off >>= 1) {
    if (tid < off) { rs_[tid] += rs_[tid + off]; rq[tid] += rq[tid + off]; }
    __syncthreads();
  }
  mu = rs_[0] * (1.f / HH);
  var = rq[0] * (1.f / HH) - mu * mu;
  rstd = rsqrtf(var + 1e-12f);
  g4 = *(const float4*)(g2 + c); bb4 = *(const float4*)(b2 + c);
  ushort4 o;
  o.x = f2bf((a0 - mu) * rstd * g4.x + bb4.x);
  o.y = f2bf((a1 - mu) * rstd * g4.y + bb4.y);
  o.z = f2bf((a2 - mu) * rstd * g4.z + bb4.z);
  o.w = f2bf((a3 - mu) * rstd * g4.w + bb4.w);
  *(ushort4*)(xln + (size_t)row * HH + c) = o;
}

// ---------------- router: colsum over S, then logits + argmax ----------------
__global__ __launch_bounds__(256) void colsum_kernel(
    const unsigned short* __restrict__ xln, float* __restrict__ colsum) {
  int b = blockIdx.x >> 2;
  int h = ((blockIdx.x & 3) << 8) + threadIdx.x;
  const unsigned short* p = xln + (size_t)b * SS * HH + h;
  float s = 0.f;
  for (int i = 0; i < SS; ++i) s += bf2f(p[(size_t)i * HH]);
  colsum[b * HH + h] = s;
}

__global__ __launch_bounds__(256) void router_kernel(
    const float* __restrict__ colsum, const float* __restrict__ gateW,
    float* __restrict__ logits_out, int* __restrict__ choice) {
  __shared__ float red[256];
  __shared__ float lg[EE];
  int b = blockIdx.x, tid = threadIdx.x;
  int e = tid & 7, part = tid >> 3;  // 32 partials per expert
  float s = 0.f;
  for (int h0 = part; h0 < HH; h0 += 32) s += colsum[b * HH + h0] * gateW[h0 * EE + e];
  red[tid] = s;
  __syncthreads();
  for (int off = 128; off >= 8; off >>= 1) {
    if (tid < off) red[tid] += red[tid + off];
    __syncthreads();
  }
  if (tid < EE) {
    float L = red[tid] * (1.0f / SS);
    logits_out[b * EE + tid] = L;
    lg[tid] = L;
  }
  __syncthreads();
  if (tid == 0) {
    int best = 0; float bv = lg[0];
    for (int e2 = 1; e2 < EE; ++e2)
      if (lg[e2] > bv) { bv = lg[e2]; best = e2; }
    choice[b] = best;
  }
}

// ---------------- host launcher ----------------
extern "C" void kernel_launch(void* const* d_in, const int* in_sizes, int n_in,
                              void* d_out, int out_size, void* d_ws, size_t ws_size,
                              hipStream_t stream) {
  const float* x     = (const float*)d_in[0];
  const float* Wq    = (const float*)d_in[1];
  const float* bq    = (const float*)d_in[2];
  const float* Wk    = (const float*)d_in[3];
  const float* bk    = (const float*)d_in[4];
  const float* Wv    = (const float*)d_in[5];
  const float* bv    = (const float*)d_in[6];
  const float* Wo    = (const float*)d_in[7];
  const float* bo    = (const float*)d_in[8];
  const float* ln1g  = (const float*)d_in[9];
  const float* ln1b  = (const float*)d_in[10];
  const float* ln2g  = (const float*)d_in[11];
  const float* ln2b  = (const float*)d_in[12];
  const float* gateW = (const float*)d_in[13];
  const float* W_up  = (const float*)d_in[14];
  const float* b_up  = (const float*)d_in[15];
  const float* W_dn  = (const float*)d_in[16];
  const float* b_dn  = (const float*)d_in[17];

  char* ws = (char*)d_ws;
  constexpr size_t QB = (size_t)BB * SS * HH * 2;  // one bf16 [B*S,H] buffer = 16.78 MB
  // lifetime-based layout (~117.5 MB total):
  unsigned short* q    = (unsigned short*)(ws + 0 * QB);
  unsigned short* kbuf = (unsigned short*)(ws + 1 * QB);
  unsigned short* vbuf = (unsigned short*)(ws + 2 * QB);
  unsigned short* ctx  = (unsigned short*)(ws + 3 * QB);
  float* y             = (float*)(ws + 0 * QB);  // reuses q+k after attention
  float* attn_out      = (float*)(ws + 4 * QB);  // spans 2*QB, lives to the end
  unsigned short* xln  = (unsigned short*)(ws + 6 * QB);
  unsigned short* hbuf = (unsigned short*)(ws + 0 * QB);  // 4*QB, reuses q..ctx
  float* colsum        = (float*)(ws + 7 * QB);
  int* choice          = (int*)(ws + 7 * QB + 65536);

  dim3 blk(256);
  const long long zero = 0;

  // QKV projections: [8192,1024] x [1024,1024] -> bf16
  gemm_wmma<false, EPI_BF16_BIAS><<<dim3(8, 64, 1), blk, 0, stream>>>(
      x, Wq, bq, nullptr, q, BB * SS, HH, HH, nullptr, zero, zero, zero, zero, zero);
  gemm_wmma<false, EPI_BF16_BIAS><<<dim3(8, 64, 1), blk, 0, stream>>>(
      x, Wk, bk, nullptr, kbuf, BB * SS, HH, HH, nullptr, zero, zero, zero, zero, zero);
  gemm_wmma<false, EPI_BF16_BIAS><<<dim3(8, 64, 1), blk, 0, stream>>>(
      x, Wv, bv, nullptr, vbuf, BB * SS, HH, HH, nullptr, zero, zero, zero, zero, zero);

  // flash attention per (b,h, 128 q-rows)
  attn_kernel<<<dim3(BB * NHH, SS / 128, 1), blk, 0, stream>>>(q, kbuf, vbuf, ctx);

  // output projection + bias + residual(x)
  gemm_wmma<true, EPI_F32_BIAS_RES><<<dim3(8, 64, 1), blk, 0, stream>>>(
      ctx, Wo, bo, x, y, BB * SS, HH, HH, nullptr, zero, zero, zero, zero, zero);

  // fused LN1 -> attn_out (f32), LN2 -> x_ln (bf16)
  ln_kernel<<<dim3(BB * SS), blk, 0, stream>>>(y, ln1g, ln1b, ln2g, ln2b, attn_out, xln);

  // router -> logits tail of d_out, choice in scratch
  colsum_kernel<<<dim3(BB * 4), blk, 0, stream>>>(xln, colsum);
  router_kernel<<<dim3(BB), blk, 0, stream>>>(
      colsum, gateW, (float*)d_out + (size_t)BB * SS * HH, choice);

  // MoE up: per-batch expert GEMM [512,1024]x[1024,4096] + GELU -> bf16 h
  gemm_wmma<true, EPI_GELU_BF16><<<dim3(II / 128, SS / 128, BB), blk, 0, stream>>>(
      xln, W_up, b_up, nullptr, hbuf, SS, II, HH, choice,
      (long long)SS * HH, (long long)HH * II, (long long)II, zero, (long long)SS * II);

  // MoE down: [512,4096]x[4096,1024] + bias + residual(attn_out) -> d_out
  gemm_wmma<true, EPI_F32_BIAS_RES><<<dim3(HH / 128, SS / 128, BB), blk, 0, stream>>>(
      hbuf, W_dn, b_dn, attn_out, d_out, SS, HH, II, choice,
      (long long)SS * II, (long long)II * HH, (long long)HH,
      (long long)SS * HH, (long long)SS * HH);
}
